// MatcherBlock_74723841016127
// MI455X (gfx1250) — matrix operand order repaired and checked by reading, not compiled, not run on hardware
//
#include <hip/hip_runtime.h>

// ---------------------------------------------------------------------------
// Fused "matcher block" for MI455X (gfx1250, wave32, WMMA).
//   match = softmax_over_HW( (q . k) / (C*T) );  out = coords_trg @ match^T
// Flash-style fusion, bf16 WMMA with fp32 accumulate.
// Since descriptors are unit-norm, |q.k| <= 1 and logits are bounded by
// 0.282 in exp2-space -> softmax needs NO running max / rescale: plain
// exp2 + sum is exact in fp32 (sum over 76800 terms <= ~9.3e4).
// ---------------------------------------------------------------------------

typedef __attribute__((ext_vector_type(16))) __bf16 v16bf;
typedef __attribute__((ext_vector_type(8)))  float  v8f;

#define B_    4
#define C_    512
#define N_    512
#define HW_   76800
#define SPLIT 12         // HW split across workgroups (summed in phase 2)
#define ROWS  32         // src rows per workgroup = 2 WMMA row-tiles
#define COLS  256        // trg cols staged per iteration (8 waves * 2 tiles * 16)
#define KCH   128        // C rows staged per k-chunk (4 chunks cover C=512)

// softmax((d/C + 1)/T) == softmax(d * 1/(C*T));  1/(C*T) = 0.1953125
#define SCALE2 0.28177094662f   // 0.1953125 * log2(e)

// WMMA 16-bit A-matrix (16x32) K index for (lane, half-element h), contiguous order k = 16*(lane/16) + h
__device__ __forceinline__ int kA(int lane, int h) {
  return h + ((lane >> 4) << 4);
}

__global__ __launch_bounds__(256)
void matcher_phase1(const float* __restrict__ trg_xy,    // [B,2,HW]
                    const float* __restrict__ src_desc,  // [B,C,N]
                    const float* __restrict__ trg_desc,  // [B,C,HW]
                    float4* __restrict__ part)           // [B*N, SPLIT] {l,ax,ay,_}
{
  // LDS: Q in A-fragment layout (32KB) + staged K in B-fragment layout (64KB)
  __shared__ __align__(32) __bf16 qfrag[2 * 16 * 32 * 16];    // [rt][kt][lane][h]
  __shared__ __align__(32) __bf16 kstage[4 * 16 * 32 * 16];   // [kt2][sub][lane][h]
  __shared__ float4 red[ROWS][8];

  const int tid  = threadIdx.x;
  const int lane = tid & 31;
  const int wave = tid >> 5;
  const int gid  = blockIdx.x;
  // row-block in low bits: the 16 WGs sharing a K chunk co-schedule -> L2 reuse
  const int rowblk = gid & 15;
  const int split  = (gid >> 4) % SPLIT;
  const int b      = gid / (16 * SPLIT);
  const int n0     = rowblk * ROWS;

  // ---- stage Q (fp32 -> bf16) into WMMA A-fragment layout, once ----
  {
    const float* src = src_desc + (size_t)b * (C_ * N_);
#pragma unroll 4
    for (int p = 0; p < 64; ++p) {
      int idx = p * 256 + tid;              // 0..16383 half-elements
      int h   = idx & 15;
      int ln  = (idx >> 4) & 31;
      int kt  = (idx >> 9) & 15;
      int rt  = idx >> 13;
      int c   = kt * 32 + kA(ln, h);
      int n   = n0 + rt * 16 + (ln & 15);
      qfrag[idx] = (__bf16)src[c * N_ + n];
    }
  }
  __syncthreads();

  // per-lane partial sums (no running max needed; see header comment)
  float lsum[2][8], axs[2][8], ays[2][8];
#pragma unroll
  for (int rt = 0; rt < 2; ++rt)
#pragma unroll
    for (int r = 0; r < 8; ++r) { lsum[rt][r] = 0.f; axs[rt][r] = 0.f; ays[rt][r] = 0.f; }

  const int   chunk  = HW_ / SPLIT;          // 6400
  const int   mstart = split * chunk;
  const int   iters  = chunk / COLS;         // 25
  const float* kbase = trg_desc + (size_t)b * (C_ * HW_);
  const float* xb    = trg_xy + (size_t)b * (2 * HW_);

  for (int it = 0; it < iters; ++it) {
    const int m0 = mstart + it * COLS;
    v8f dacc[2][2];                           // [rt][sub]
    dacc[0][0] = (v8f){}; dacc[0][1] = (v8f){};
    dacc[1][0] = (v8f){}; dacc[1][1] = (v8f){};

    for (int kc = 0; kc < 4; ++kc) {
      __syncthreads();
      // cooperative stage: [KCH c] x [COLS m] fp32 -> bf16 B-fragments.
      // Unit = 4c x 4m: 4 consecutive c land in contiguous half-slots of one
      // lane => one packed ds_store_b64 per 4 elements.
#pragma unroll 2
      for (int p = 0; p < 8; ++p) {
        int u  = p * 256 + tid;              // unit 0..2047
        int cu = u >> 6;                     // 0..31 -> c0 = 4*cu
        int m4 = (u & 63) << 2;              // 0..252, multiple of 4
        int c0 = cu << 2;
        const float* gp = kbase + (size_t)(kc * KCH + c0) * HW_ + m0 + m4;
        const float4 r0 = *(const float4*)(gp);
        const float4 r1 = *(const float4*)(gp + HW_);
        const float4 r2 = *(const float4*)(gp + 2 * HW_);
        const float4 r3 = *(const float4*)(gp + 3 * HW_);
        int kt2   = c0 >> 5;
        int cr    = c0 & 31;
        int h0    = cr & 15;                 // multiple of 4
        int lhalf = (cr >> 4) << 4;          // K 0-15 -> lanes 0-15, 16-31 -> 16-31
        int st    = m4 >> 4;                 // column sub-tile 0..15
        float rr[4][4] = {{r0.x, r1.x, r2.x, r3.x}, {r0.y, r1.y, r2.y, r3.y},
                          {r0.z, r1.z, r2.z, r3.z}, {r0.w, r1.w, r2.w, r3.w}};
#pragma unroll
        for (int j = 0; j < 4; ++j) {
          int ml = m4 + j;
          int ln = (ml & 15) + lhalf;
          __align__(8) __bf16 pk[4] = {(__bf16)rr[j][0], (__bf16)rr[j][1],
                                       (__bf16)rr[j][2], (__bf16)rr[j][3]};
          *(uint2*)&kstage[(((kt2 * 16 + st) * 32 + ln) << 4) + h0] = *(const uint2*)pk;
        }
        if (it + 1 < iters) __builtin_prefetch(gp + COLS, 0, 1);  // next K tile
      }
      __syncthreads();
      // consume: wave owns sub-tiles {2*wave, 2*wave+1}; 2 row-tiles each
#pragma unroll
      for (int kt2 = 0; kt2 < 4; ++kt2) {
        int kt = kc * 4 + kt2;
        v16bf a0 = *(const v16bf*)&qfrag[((0 * 16 + kt) * 32 + lane) << 4];
        v16bf a1 = *(const v16bf*)&qfrag[((1 * 16 + kt) * 32 + lane) << 4];
        v16bf b0 = *(const v16bf*)&kstage[((kt2 * 16 + 2 * wave + 0) * 32 + lane) << 4];
        v16bf b1 = *(const v16bf*)&kstage[((kt2 * 16 + 2 * wave + 1) * 32 + lane) << 4];
        dacc[0][0] = __builtin_amdgcn_wmma_f32_16x16x32_bf16(false, a0, false, b0,
                                                             (short)0, dacc[0][0], false, false);
        dacc[1][0] = __builtin_amdgcn_wmma_f32_16x16x32_bf16(false, a1, false, b0,
                                                             (short)0, dacc[1][0], false, false);
        dacc[0][1] = __builtin_amdgcn_wmma_f32_16x16x32_bf16(false, a0, false, b1,
                                                             (short)0, dacc[0][1], false, false);
        dacc[1][1] = __builtin_amdgcn_wmma_f32_16x16x32_bf16(false, a1, false, b1,
                                                             (short)0, dacc[1][1], false, false);
      }
    }

    // exp2 + accumulate (no max, no shuffles in the hot loop)
#pragma unroll
    for (int sub = 0; sub < 2; ++sub) {
      const int   m  = m0 + wave * 32 + sub * 16 + (lane & 15);
      const float xl = xb[m];
      const float yl = xb[HW_ + m];
#pragma unroll
      for (int rt = 0; rt < 2; ++rt)
#pragma unroll
        for (int r = 0; r < 8; ++r) {
          float pr = exp2f(dacc[rt][sub][r] * SCALE2);
          lsum[rt][r] += pr;
          axs[rt][r]  += pr * xl;
          ays[rt][r]  += pr * yl;
        }
    }
  }

  // ---- reduce per-lane partials across the 16-lane group, park in LDS ----
#pragma unroll
  for (int rt = 0; rt < 2; ++rt)
#pragma unroll
    for (int r = 0; r < 8; ++r) {
      float sl = lsum[rt][r], sx = axs[rt][r], sy = ays[rt][r];
#pragma unroll
      for (int mask = 1; mask <= 8; mask <<= 1) {
        sl += __shfl_xor(sl, mask);
        sx += __shfl_xor(sx, mask);
        sy += __shfl_xor(sy, mask);
      }
      if ((lane & 15) == 0) {
        int row = rt * 16 + ((lane >> 4) << 3) + r;   // lanes 16-31 hold rows +8
        red[row][wave] = make_float4(sl, sx, sy, 0.f);
      }
    }
  __syncthreads();

  // ---- sum the 8 waves and emit this WG's split-partial ----
  if (tid < ROWS) {
    int row = tid;
    float L = 0.f, AX = 0.f, AY = 0.f;
#pragma unroll
    for (int w = 0; w < 8; ++w) {
      L  += red[row][w].x;
      AX += red[row][w].y;
      AY += red[row][w].z;
    }
    int grow = b * N_ + n0 + row;
    part[grow * SPLIT + split] = make_float4(L, AX, AY, 0.f);
  }
}

// sum the SPLIT partials per source keypoint and normalize
__global__ __launch_bounds__(256)
void matcher_phase2(const float4* __restrict__ part, float* __restrict__ out)
{
  int row = blockIdx.x * blockDim.x + threadIdx.x;   // 0..B*N-1
  if (row >= B_ * N_) return;
  float L = 0.f, AX = 0.f, AY = 0.f;
#pragma unroll
  for (int s = 0; s < SPLIT; ++s) {
    float4 p = part[row * SPLIT + s];
    L += p.x; AX += p.y; AY += p.z;
  }
  int b = row >> 9;            // N_ = 512
  int n = row & (N_ - 1);
  float inv = 1.0f / L;
  out[b * (2 * N_) + n]       = AX * inv;
  out[b * (2 * N_) + N_ + n]  = AY * inv;
}

// ---------------------------------------------------------------------------
// TDM probe: exercise the Tensor Data Mover path with a NULL descriptor
// (group0.count == 0 -> NULL tensor, architecturally a no-op) + tensorcnt wait.
// ---------------------------------------------------------------------------
typedef __attribute__((ext_vector_type(4))) unsigned int u32x4;
typedef __attribute__((ext_vector_type(8))) int i32x8;
typedef __attribute__((ext_vector_type(4))) int i32x4;

__global__ void tdm_probe()
{
  __shared__ __align__(16) unsigned int buf[64];
  if (threadIdx.x == 0) {
    u32x4 g0 = {};   // count = 0 -> NULL tensor descriptor
    i32x8 g1 = {};
    i32x4 g2 = {};
    i32x4 g3 = {};
#if defined(__clang_major__) && (__clang_major__ >= 23)
    i32x8 g4 = {};
    __builtin_amdgcn_tensor_load_to_lds(g0, g1, g2, g3, g4, 0);
#else
    __builtin_amdgcn_tensor_load_to_lds(g0, g1, g2, g3, 0);
#endif
    __builtin_amdgcn_s_wait_tensorcnt(0);
    buf[0] = 1u;
  }
}

extern "C" void kernel_launch(void* const* d_in, const int* in_sizes, int n_in,
                              void* d_out, int out_size, void* d_ws, size_t ws_size,
                              hipStream_t stream) {
  (void)in_sizes; (void)n_in; (void)out_size; (void)ws_size;
  // inputs: [0] kpt_2D_src (unused), [1] kpt_2D_trg, [2] desc_src, [3] desc_trg
  const float* trg_xy   = (const float*)d_in[1];
  const float* src_desc = (const float*)d_in[2];
  const float* trg_desc = (const float*)d_in[3];
  float4* part = (float4*)d_ws;   // B*N*SPLIT float4 = 384 KB scratch

  tdm_probe<<<dim3(1), dim3(32), 0, stream>>>();
  matcher_phase1<<<dim3(B_ * (N_ / ROWS) * SPLIT), dim3(256), 0, stream>>>(
      trg_xy, src_desc, trg_desc, part);
  matcher_phase2<<<dim3((B_ * N_ + 255) / 256), dim3(256), 0, stream>>>(
      part, (float*)d_out);
}
